// Patient2Vec_51737176048064
// MI455X (gfx1250) — matrix operand order repaired and checked by reading, not compiled, not run on hardware
//
#include <hip/hip_runtime.h>
#include <math.h>

typedef __attribute__((ext_vector_type(16))) __bf16 v16bf;
typedef __attribute__((ext_vector_type(8)))  __bf16 v8bf;
typedef __attribute__((ext_vector_type(8)))  float  v8f;

constexpr int B_ = 64, S_ = 64, P_ = 32, D_ = 512, E_ = 512, H_ = 512;
constexpr int ATT_ = 256, NH_ = 4, OUT_ = 16;
constexpr int BS_ = B_ * S_;        // 4096
constexpr int G3_ = 3 * H_;         // 1536
constexpr int H2_ = 2 * H_;         // 1024
constexpr int BH_ = B_ * H_;        // 32768

// d_out layout (floats): out | states | context | alpha_actv
constexpr size_t OFF_OUT    = 0;
constexpr size_t OFF_STATES = (size_t)B_ * OUT_;                        // 1024
constexpr size_t OFF_CTX    = OFF_STATES + (size_t)BS_ * H2_;           // 4195328
constexpr size_t OFF_ALPHA  = OFF_CTX + (size_t)B_ * NH_ * H2_;         // 4457472

// ---------------- WMMA fragment loaders (bf16, 16x16x32) ----------------
// A: 16(M) x 32(K). lane&15 = M row; lanes 0-15 get K {half*8+i / 16+half*8+i}.
__device__ __forceinline__ v16bf load_fragA(const __bf16* tile, int ldk, int lane) {
  const int r = lane & 15, half = lane >> 4;
  const __bf16* p = tile + (size_t)r * ldk + half * 8;
  v8bf lo = *reinterpret_cast<const v8bf*>(p);
  v8bf hi = *reinterpret_cast<const v8bf*>(p + 16);
  v16bf o;
#pragma unroll
  for (int i = 0; i < 8; ++i) { o[i] = lo[i]; o[i + 8] = hi[i]; }
  return o;
}
// B: 32(K) x 16(N), stored [N,K] row-major. lane&15 = N col; lanes hold K {half*8+i / 16+half*8+i}.
__device__ __forceinline__ v16bf load_fragB(const __bf16* tile, int ldk, int lane) {
  const int n = lane & 15, half = lane >> 4;
  const __bf16* p = tile + (size_t)n * ldk + half * 8;
  v8bf lo = *reinterpret_cast<const v8bf*>(p);
  v8bf hi = *reinterpret_cast<const v8bf*>(p + 16);
  v16bf o;
#pragma unroll
  for (int i = 0; i < 8; ++i) { o[i] = lo[i]; o[i + 8] = hi[i]; }
  return o;
}

__device__ __forceinline__ float sigmoidf_(float x) { return 1.f / (1.f + expf(-x)); }

// ---------------- f32 -> bf16 conversion ----------------
__global__ void cvt_bf16_k(const float* __restrict__ src, __bf16* __restrict__ dst, int n) {
  int i = blockIdx.x * blockDim.x + threadIdx.x;
  if (i < n) dst[i] = (__bf16)src[i];
}

// ---------------- fused conv + weighted sum: vec[b,s,d] (bf16) ----------------
__global__ void conv_vec_k(const float* __restrict__ in, const float* __restrict__ conv_w,
                           const float* __restrict__ conv_b, __bf16* __restrict__ vecB) {
  __shared__ float cw[D_];
  __shared__ float cv[P_];
  const int tid = threadIdx.x;
  const size_t base = (size_t)blockIdx.x * P_ * D_;
  cw[tid] = conv_w[tid];
  cw[tid + 256] = conv_w[tid + 256];
  __syncthreads();
  const int lane = tid & 31, wave = tid >> 5;
#pragma unroll
  for (int pp = 0; pp < 4; ++pp) {
    const int p = wave * 4 + pp;
    const float* row = in + base + (size_t)p * D_;
    float s = 0.f;
    for (int d = lane; d < D_; d += 32) s += row[d] * cw[d];
#pragma unroll
    for (int o = 16; o > 0; o >>= 1) s += __shfl_xor(s, o, 32);
    if (lane == 0) cv[p] = s + conv_b[0];
  }
  __syncthreads();
#pragma unroll
  for (int q = 0; q < 2; ++q) {
    const int d = tid + q * 256;
    float acc = 0.f;
    for (int p = 0; p < P_; ++p) acc += cv[p] * in[base + (size_t)p * D_ + d];
    vecB[(size_t)blockIdx.x * D_ + d] = (__bf16)acc;
  }
}

// ---------------- generic WMMA GEMM: C[M,N] = A[M,K] * B[N,K]^T (+bias)(clamp) ----------------
template <bool CLAMP, bool HAS_BIAS, bool OUT_F32, bool OUT_BF16, bool PERMUTE_SB>
__global__ void gemm_wmma_k(const __bf16* __restrict__ Amat, const __bf16* __restrict__ Bmat,
                            const float* __restrict__ bias, float* __restrict__ outF,
                            __bf16* __restrict__ outB, int M, int N, int K) {
  const int lane = threadIdx.x & 31;
  const int wave = threadIdx.x >> 5;
  const int ntn = N >> 4;
  const int tile = blockIdx.x * 8 + wave;
  const int m0 = (tile / ntn) << 4;
  const int n0 = (tile % ntn) << 4;
  if (m0 >= M) return;
  v8f acc = {};
  for (int k0 = 0; k0 < K; k0 += 32) {
    v16bf a = load_fragA(Amat + (size_t)m0 * K + k0, K, lane);
    v16bf b = load_fragB(Bmat + (size_t)n0 * K + k0, K, lane);
    acc = __builtin_amdgcn_wmma_f32_16x16x32_bf16(false, a, false, b, (short)0, acc, false, false);
  }
  const int half = lane >> 4;
  const int col = n0 + (lane & 15);
  const float bv = HAS_BIAS ? bias[col] : 0.f;
#pragma unroll
  for (int r = 0; r < 8; ++r) {
    const int mrow = m0 + r + 8 * half;
    float v = acc[r] + bv;
    if (CLAMP) v = fminf(1.f, fmaxf(-1.f, v));
    int orow = mrow;
    if (PERMUTE_SB) { int bb = mrow / S_; int ss = mrow - bb * S_; orow = ss * B_ + bb; }
    if (OUT_F32) outF[(size_t)orow * N + col] = v;
    if (OUT_BF16) outB[(size_t)orow * N + col] = (__bf16)v;
  }
}

// ---------------- zero initial hidden state (both directions, ping slot 0) ----------------
__global__ void init_h_k(float* hfF, __bf16* hbF, float* hfB, __bf16* hbB) {
  int i = blockIdx.x * blockDim.x + threadIdx.x;
  if (i < BH_) { hfF[i] = 0.f; hfB[i] = 0.f; hbF[i] = (__bf16)0.f; hbB[i] = (__bf16)0.f; }
}

// ---------------- one GRU time step, both directions (blockIdx.z) ----------------
// Each wave: 16 batch rows x 16 hidden cols, three WMMA chains (r,z,n) over K=512.
__global__ void gru_step_k(const __bf16* hbF_in, const float* hfF_in, __bf16* hbF_out, float* hfF_out,
                           const __bf16* hbB_in, const float* hfB_in, __bf16* hbB_out, float* hfB_out,
                           const float* __restrict__ xgF, const float* __restrict__ xgB,
                           const __bf16* __restrict__ wHhF, const __bf16* __restrict__ wHhB,
                           const float* __restrict__ bHhF, const float* __restrict__ bHhB,
                           float* __restrict__ statesF, __bf16* __restrict__ statesB, int t) {
  const int dir = blockIdx.z;
  const __bf16* h_in_b = dir ? hbB_in : hbF_in;
  const float*  h_in_f = dir ? hfB_in : hfF_in;
  __bf16* h_out_b = dir ? hbB_out : hbF_out;
  float*  h_out_f = dir ? hfB_out : hfF_out;
  const float*  xg  = dir ? xgB : xgF;
  const __bf16* whh = dir ? wHhB : wHhF;
  const float*  bhh = dir ? bHhB : bHhF;
  const int s = dir ? (S_ - 1 - t) : t;

  const int lane = threadIdx.x & 31;
  const int wave = threadIdx.x >> 5;
  const int tile = blockIdx.x * 8 + wave;   // 0..127
  const int m0 = (tile >> 5) << 4;          // batch tile
  const int j0 = (tile & 31) << 4;          // hidden tile

  v8f ar = {}, az = {}, an = {};
  for (int k0 = 0; k0 < H_; k0 += 32) {
    v16bf a  = load_fragA(h_in_b + (size_t)m0 * H_ + k0, H_, lane);
    v16bf br = load_fragB(whh + (size_t)(j0)*H_ + k0, H_, lane);
    v16bf bz = load_fragB(whh + (size_t)(j0 + H_) * H_ + k0, H_, lane);
    v16bf bn = load_fragB(whh + (size_t)(j0 + 2 * H_) * H_ + k0, H_, lane);
    ar = __builtin_amdgcn_wmma_f32_16x16x32_bf16(false, a, false, br, (short)0, ar, false, false);
    az = __builtin_amdgcn_wmma_f32_16x16x32_bf16(false, a, false, bz, (short)0, az, false, false);
    an = __builtin_amdgcn_wmma_f32_16x16x32_bf16(false, a, false, bn, (short)0, an, false, false);
  }
  const int half = lane >> 4;
  const int j = j0 + (lane & 15);
  const float bhr = bhh[j], bhz = bhh[H_ + j], bhn = bhh[2 * H_ + j];
#pragma unroll
  for (int r = 0; r < 8; ++r) {
    const int b = m0 + r + 8 * half;
    const size_t xrow = ((size_t)s * B_ + b) * G3_;
    const float hr = ar[r] + bhr, hz = az[r] + bhz, hn = an[r] + bhn;
    const float xr = xg[xrow + j], xz = xg[xrow + H_ + j], xn = xg[xrow + 2 * H_ + j];
    const float rg = sigmoidf_(xr + hr);
    const float zg = sigmoidf_(xz + hz);
    const float ng = tanhf(xn + rg * hn);
    const float hp = h_in_f[(size_t)b * H_ + j];
    const float hnew = (1.f - zg) * ng + zg * hp;
    h_out_f[(size_t)b * H_ + j] = hnew;
    h_out_b[(size_t)b * H_ + j] = (__bf16)hnew;
    const size_t srow = ((size_t)b * S_ + s) * H2_ + (size_t)dir * H_ + j;
    statesF[srow] = hnew;
    statesB[srow] = (__bf16)hnew;
  }
}

// ---------------- alpha = m1 @ att_w2^T  (N=4 < tile -> plain VALU) ----------------
__global__ void alpha_k(const __bf16* __restrict__ m1, const float* __restrict__ att_w2,
                        float* __restrict__ alpha) {
  const int idx = blockIdx.x * blockDim.x + threadIdx.x;
  if (idx >= BS_ * NH_) return;
  const int row = idx >> 2, h = idx & 3;
  const __bf16* mr = m1 + (size_t)row * ATT_;
  const float* w = att_w2 + (size_t)h * ATT_;
  float s = 0.f;
  for (int a = 0; a < ATT_; ++a) s += (float)mr[a] * w[a];
  alpha[idx] = s;   // [(b*S+s)][h]
}

// ---------------- softmax over S per (b,h): alpha_actv[b,h,s] ----------------
__global__ void softmax_alpha_k(const float* __restrict__ alpha, float* __restrict__ alpha_out) {
  __shared__ float red[S_];
  const int b = blockIdx.x >> 2, h = blockIdx.x & 3;
  const int s = threadIdx.x;
  const float v = alpha[((size_t)b * S_ + s) * NH_ + h];
  red[s] = v; __syncthreads();
  for (int o = 32; o > 0; o >>= 1) { if (s < o) red[s] = fmaxf(red[s], red[s + o]); __syncthreads(); }
  const float mx = red[0]; __syncthreads();
  const float e = expf(v - mx);
  red[s] = e; __syncthreads();
  for (int o = 32; o > 0; o >>= 1) { if (s < o) red[s] += red[s + o]; __syncthreads(); }
  alpha_out[(size_t)b * NH_ * S_ + (size_t)h * S_ + s] = e / red[0];
}

// ---------------- context[b,h,:] = sum_s alpha_actv[b,h,s] * states[b,s,:] ----------------
__global__ void context_k(const float* __restrict__ alpha_actv, const float* __restrict__ statesF,
                          float* __restrict__ ctx) {
  const int b = blockIdx.x >> 2, h = blockIdx.x & 3;
  const float* aw = alpha_actv + (size_t)b * NH_ * S_ + (size_t)h * S_;
  float acc[4] = {0.f, 0.f, 0.f, 0.f};
  for (int s = 0; s < S_; ++s) {
    const float w = aw[s];
    const float* st = statesF + ((size_t)b * S_ + s) * H2_;
#pragma unroll
    for (int q = 0; q < 4; ++q) acc[q] += w * st[threadIdx.x + q * 256];
  }
#pragma unroll
  for (int q = 0; q < 4; ++q)
    ctx[(size_t)b * (NH_ * H2_) + (size_t)h * H2_ + threadIdx.x + q * 256] = acc[q];
}

// ---------------- final linear (64x4096 @ 4096x16) + softmax ----------------
__global__ void lin_k(const float* __restrict__ ctx, const float* __restrict__ lin_w,
                      const float* __restrict__ lin_b, float* __restrict__ out) {
  const int b = blockIdx.x, lane = threadIdx.x;
  float v = -3.0e38f;
  if (lane < OUT_) {
    float s = lin_b[lane];
    const float* c = ctx + (size_t)b * (NH_ * H2_);
    const float* w = lin_w + (size_t)lane * (NH_ * H2_);
    for (int k = 0; k < NH_ * H2_; ++k) s += c[k] * w[k];
    v = s;
  }
  float mx = v;
  for (int o = 8; o > 0; o >>= 1) mx = fmaxf(mx, __shfl_xor(mx, o, 16));
  const float e = (lane < OUT_) ? expf(v - mx) : 0.f;
  float sum = e;
  for (int o = 8; o > 0; o >>= 1) sum += __shfl_xor(sum, o, 16);
  if (lane < OUT_) out[(size_t)b * OUT_ + lane] = e / sum;
}

// ==========================================================================
extern "C" void kernel_launch(void* const* d_in, const int* in_sizes, int n_in,
                              void* d_out, int out_size, void* d_ws, size_t ws_size,
                              hipStream_t stream) {
  (void)in_sizes; (void)n_in; (void)out_size; (void)ws_size;
  const float* inputs = (const float*)d_in[0];
  // d_in[1] = batch_size (int scalar), unused
  const float* conv_w = (const float*)d_in[2];
  const float* conv_b = (const float*)d_in[3];
  const float* w_embed = (const float*)d_in[4];
  const float* w_ih_f = (const float*)d_in[5];
  const float* w_hh_f = (const float*)d_in[6];
  const float* b_ih_f = (const float*)d_in[7];
  const float* b_hh_f = (const float*)d_in[8];
  const float* w_ih_b = (const float*)d_in[9];
  const float* w_hh_b = (const float*)d_in[10];
  const float* b_ih_b = (const float*)d_in[11];
  const float* b_hh_b = (const float*)d_in[12];
  const float* att_w1 = (const float*)d_in[13];
  const float* att_w2 = (const float*)d_in[14];
  const float* lin_w = (const float*)d_in[15];
  const float* lin_b = (const float*)d_in[16];
  float* out = (float*)d_out;

  // bump allocator over workspace (256B aligned)
  char* ws = (char*)d_ws;
  size_t off = 0;
  auto alloc = [&](size_t bytes) -> void* {
    off = (off + 255) & ~(size_t)255;
    void* p = ws + off;
    off += bytes;
    return p;
  };
  __bf16* wEmb_b   = (__bf16*)alloc((size_t)E_ * D_ * 2);
  __bf16* wIhF_b   = (__bf16*)alloc((size_t)G3_ * E_ * 2);
  __bf16* wIhB_b   = (__bf16*)alloc((size_t)G3_ * E_ * 2);
  __bf16* wHhF_b   = (__bf16*)alloc((size_t)G3_ * H_ * 2);
  __bf16* wHhB_b   = (__bf16*)alloc((size_t)G3_ * H_ * 2);
  __bf16* attW1_b  = (__bf16*)alloc((size_t)ATT_ * H2_ * 2);
  __bf16* vec_b    = (__bf16*)alloc((size_t)BS_ * D_ * 2);
  __bf16* emb_b    = (__bf16*)alloc((size_t)BS_ * E_ * 2);
  float*  xgF      = (float*) alloc((size_t)BS_ * G3_ * 4);
  float*  xgB      = (float*) alloc((size_t)BS_ * G3_ * 4);
  __bf16* states_b = (__bf16*)alloc((size_t)BS_ * H2_ * 2);
  __bf16* m1_b     = (__bf16*)alloc((size_t)BS_ * ATT_ * 2);
  float*  alphaF   = (float*) alloc((size_t)BS_ * NH_ * 4);
  float*  hfF      = (float*) alloc((size_t)2 * BH_ * 4);   // ping-pong
  float*  hfB      = (float*) alloc((size_t)2 * BH_ * 4);
  __bf16* hbF      = (__bf16*)alloc((size_t)2 * BH_ * 2);
  __bf16* hbB      = (__bf16*)alloc((size_t)2 * BH_ * 2);

  // 1) weights -> bf16
  auto cvt = [&](const float* s, __bf16* d, int n) {
    cvt_bf16_k<<<(n + 255) / 256, 256, 0, stream>>>(s, d, n);
  };
  cvt(w_embed, wEmb_b, E_ * D_);
  cvt(w_ih_f, wIhF_b, G3_ * E_);
  cvt(w_ih_b, wIhB_b, G3_ * E_);
  cvt(w_hh_f, wHhF_b, G3_ * H_);
  cvt(w_hh_b, wHhB_b, G3_ * H_);
  cvt(att_w1, attW1_b, ATT_ * H2_);

  // 2) fused conv + weighted sum (bandwidth-bound stage)
  conv_vec_k<<<BS_, 256, 0, stream>>>(inputs, conv_w, conv_b, vec_b);

  // 3) emb = clip(vec @ w_embed^T)  [4096x512x512], bf16 out
  gemm_wmma_k<true, false, false, true, false>
      <<<(BS_ / 16) * (E_ / 16) / 8, 256, 0, stream>>>(vec_b, wEmb_b, nullptr, nullptr, emb_b, BS_, E_, D_);

  // 4) xg = emb @ w_ih^T + b_ih  [4096x1536x512], f32 out in [S,B,3H] order
  gemm_wmma_k<false, true, true, false, true>
      <<<(BS_ / 16) * (G3_ / 16) / 8, 256, 0, stream>>>(emb_b, wIhF_b, b_ih_f, xgF, nullptr, BS_, G3_, E_);
  gemm_wmma_k<false, true, true, false, true>
      <<<(BS_ / 16) * (G3_ / 16) / 8, 256, 0, stream>>>(emb_b, wIhB_b, b_ih_b, xgB, nullptr, BS_, G3_, E_);

  // 5) bidirectional GRU scan (64 steps, both directions per launch)
  init_h_k<<<(BH_ + 255) / 256, 256, 0, stream>>>(hfF, hbF, hfB, hbB);
  for (int t = 0; t < S_; ++t) {
    const int pi = t & 1, po = pi ^ 1;
    gru_step_k<<<dim3(16, 1, 2), 256, 0, stream>>>(
        hbF + (size_t)pi * BH_, hfF + (size_t)pi * BH_, hbF + (size_t)po * BH_, hfF + (size_t)po * BH_,
        hbB + (size_t)pi * BH_, hfB + (size_t)pi * BH_, hbB + (size_t)po * BH_, hfB + (size_t)po * BH_,
        xgF, xgB, wHhF_b, wHhB_b, b_hh_f, b_hh_b, out + OFF_STATES, states_b, t);
  }

  // 6) m1 = clip(states @ att_w1^T)  [4096x256x1024], bf16 out
  gemm_wmma_k<true, false, false, true, false>
      <<<(BS_ / 16) * (ATT_ / 16) / 8, 256, 0, stream>>>(states_b, attW1_b, nullptr, nullptr, m1_b, BS_, ATT_, H2_);

  // 7) alpha, softmax over S, context, final linear + softmax
  alpha_k<<<(BS_ * NH_ + 127) / 128, 128, 0, stream>>>(m1_b, att_w2, alphaF);
  softmax_alpha_k<<<B_ * NH_, S_, 0, stream>>>(alphaF, out + OFF_ALPHA);
  context_k<<<B_ * NH_, 256, 0, stream>>>(out + OFF_ALPHA, out + OFF_STATES, out + OFF_CTX);
  lin_k<<<B_, 32, 0, stream>>>(out + OFF_CTX, lin_w, lin_b, out + OFF_OUT);
}
